// MultiHeadAttention_21062519619628
// MI455X (gfx1250) — matrix operand order repaired and checked by reading, not compiled, not run on hardware
//
#include <hip/hip_runtime.h>

// ---------------------------------------------------------------------------
// CDNA5 (gfx1250) WMMA multi-head attention
//   B=2, T=2048, C=1024, H=16, HD=64
// Pipeline:
//   1) f32 -> bf16 convert: x (row-major), W* (TRANSPOSED to [N,K])
//   2) WMMA GEMM (32x64 tile/wave, ping-pong software pipeline):
//        Q/K -> bf16 [B,H,T,64];  V -> bf16 [B,H,64,T]
//   3) WMMA flash attention: 4 waves / block share async-staged K/V LDS tiles
//      (global_load_async_to_lds_b128 + s_wait_asynccnt, double-buffered)
//      -> bf16 [B,T,H,64] (== row-major [B*T, C])
//   4) WMMA GEMM: out = attn@Wo + bo -> f32 d_out
// ---------------------------------------------------------------------------

typedef __attribute__((ext_vector_type(16))) __bf16 v16bf;
typedef __attribute__((ext_vector_type(8)))  __bf16 v8bf;
typedef __attribute__((ext_vector_type(8)))  float  v8f;

__device__ inline v8f zero8() {
  v8f z = {0.f, 0.f, 0.f, 0.f, 0.f, 0.f, 0.f, 0.f};
  return z;
}

// D = A x B + C, 16x16x32 bf16 -> f32
__device__ inline v8f wmma_bf16(v16bf a, v16bf b, v8f c) {
  return __builtin_amdgcn_wmma_f32_16x16x32_bf16(
      /*neg_a=*/false, a, /*neg_b=*/false, b,
      /*c_mod=*/(short)0, c, /*reuse_a=*/false, /*reuse_b=*/false);
}

// A fragment: 16x32 bf16 tile, p -> element (0,0), row stride sM, K stride 1.
__device__ inline v16bf load_frag_a(const __bf16* __restrict__ p, int sM,
                                    int lane) {
  const int m = lane & 15, half = lane >> 4;
  const __bf16* r = p + (size_t)m * sM;
  const v8bf lo = *(const v8bf*)(r + 8 * half);
  const v8bf hi = *(const v8bf*)(r + 16 + 8 * half);
  return __builtin_shufflevector(lo, hi, 0, 1, 2, 3, 4, 5, 6, 7, 8, 9, 10, 11,
                                 12, 13, 14, 15);
}

// B fragment: 32x16 bf16 tile, element (k,n) at n*sN + k (K stride 1).
__device__ inline v16bf load_frag_b(const __bf16* __restrict__ p, int sN,
                                    int lane) {
  const int n = lane & 15, half = lane >> 4;
  const __bf16* r = p + (size_t)n * sN;
  const v8bf lo = *(const v8bf*)(r + 8 * half);
  const v8bf hi = *(const v8bf*)(r + 16 + 8 * half);
  return __builtin_shufflevector(lo, hi, 0, 1, 2, 3, 4, 5, 6, 7, 8, 9, 10, 11,
                                 12, 13, 14, 15);
}

// Async copy 16B: global (saddr + 32-bit voffset) -> LDS, ASYNCcnt-tracked.
// GVS form (ISA 15.18.3 op 98): vdst = LDS byte address, vaddr = i32 offset.
__device__ inline void async_b128(const __bf16* lds_dst, int gbyte_off,
                                  const void* gbase) {
  const unsigned int lds_off = (unsigned int)(unsigned long long)lds_dst;
  asm volatile("global_load_async_to_lds_b128 %0, %1, %2"
               :
               : "v"(lds_off), "v"(gbyte_off),
                 "s"((unsigned long long)gbase)
               : "memory");
}

__device__ inline void wait_async0() {
  asm volatile("s_wait_asynccnt 0" ::: "memory");
}

// ---------------------------------------------------------------------------
__global__ void f32_to_bf16_kernel(const float* __restrict__ in,
                                   __bf16* __restrict__ out, int n) {
  int i = blockIdx.x * blockDim.x + threadIdx.x;
  if (i < n) out[i] = (__bf16)in[i];
}

// W [K,N] f32 row-major -> Wt [N,K] bf16 (so GEMM B-fragments are K-contiguous)
__global__ void transpose_to_bf16_kernel(const float* __restrict__ in,
                                         __bf16* __restrict__ out, int K,
                                         int N) {
  int i = blockIdx.x * blockDim.x + threadIdx.x;
  if (i < K * N) {
    int k = i / N, n = i - k * N;
    out[(size_t)n * K + k] = (__bf16)in[i];
  }
}

// ---------------------------------------------------------------------------
// WMMA GEMM: out[M,N] = A[M,K](bf16 row-major) x Bt[N,K](bf16) + bias.
// One wave computes a 32(M) x 64(N) tile. K is processed in steps of 64 with
// two ping-pong fragment buffers: loads for step k+32/k+64 are issued before
// the WMMAs that consume step k, with no register-copy rotation.
// Requires K % 64 == 0.
// mode 0: bf16 store, head-major [B, H, T, D]       (Q, K)
// mode 1: f32  store, row-major [M, N]              (final projection)
// mode 2: bf16 store, head-transposed [B, H, D, T]  (V)
__global__ __launch_bounds__(32)
void wmma_gemm_bf16(const __bf16* __restrict__ A, int lda,
                    const __bf16* __restrict__ Bt, int ldb,
                    const float* __restrict__ bias,
                    void* __restrict__ outp, int mode,
                    int M, int N, int K,
                    int Tt, int Hh, int Dd) {
  const int lane = threadIdx.x & 31;
  const int m0 = blockIdx.x * 32;
  const int n0 = blockIdx.y * 64;

  v8f acc[2][4];
#pragma unroll
  for (int mi = 0; mi < 2; ++mi)
#pragma unroll
    for (int s = 0; s < 4; ++s) acc[mi][s] = zero8();

  v16bf a0[2], b0[4], a1[2], b1[4];

#pragma unroll
  for (int mi = 0; mi < 2; ++mi)
    a0[mi] = load_frag_a(A + (size_t)(m0 + 16 * mi) * lda, lda, lane);
#pragma unroll
  for (int s = 0; s < 4; ++s)
    b0[s] = load_frag_b(Bt + (size_t)(n0 + 16 * s) * ldb, ldb, lane);

  for (int k0 = 0; k0 < K; k0 += 64) {
    // Load odd buffer (k0+32) while even WMMAs run.
    {
      const int k1 = k0 + 32;
      __builtin_prefetch(A + (size_t)m0 * lda + k0 + 64, 0, 1);
      __builtin_prefetch(Bt + (size_t)n0 * ldb + k0 + 64, 0, 1);
#pragma unroll
      for (int mi = 0; mi < 2; ++mi)
        a1[mi] = load_frag_a(A + (size_t)(m0 + 16 * mi) * lda + k1, lda, lane);
#pragma unroll
      for (int s = 0; s < 4; ++s)
        b1[s] = load_frag_b(Bt + (size_t)(n0 + 16 * s) * ldb + k1, ldb, lane);
    }
#pragma unroll
    for (int mi = 0; mi < 2; ++mi)
#pragma unroll
      for (int s = 0; s < 4; ++s)
        acc[mi][s] = wmma_bf16(a0[mi], b0[s], acc[mi][s]);

    // Load even buffer for the next K-step while odd WMMAs run.
    if (k0 + 64 < K) {
      const int k2 = k0 + 64;
#pragma unroll
      for (int mi = 0; mi < 2; ++mi)
        a0[mi] = load_frag_a(A + (size_t)(m0 + 16 * mi) * lda + k2, lda, lane);
#pragma unroll
      for (int s = 0; s < 4; ++s)
        b0[s] = load_frag_b(Bt + (size_t)(n0 + 16 * s) * ldb + k2, ldb, lane);
    }
#pragma unroll
    for (int mi = 0; mi < 2; ++mi)
#pragma unroll
      for (int s = 0; s < 4; ++s)
        acc[mi][s] = wmma_bf16(a1[mi], b1[s], acc[mi][s]);
  }

  const int nl = lane & 15, half = lane >> 4;
#pragma unroll
  for (int mi = 0; mi < 2; ++mi) {
#pragma unroll
    for (int s = 0; s < 4; ++s) {
      const int n = n0 + 16 * s + nl;
      const float bv = bias ? bias[n] : 0.f;
#pragma unroll
      for (int v = 0; v < 8; ++v) {
        const int m = m0 + 16 * mi + v + 8 * half;
        const float val = acc[mi][s][v] + bv;
        if (mode == 0) {
          const int bb = m / Tt, t = m - bb * Tt;
          const int hh = n / Dd, d = n - hh * Dd;
          ((__bf16*)outp)[(((size_t)bb * Hh + hh) * (size_t)Tt + t) * Dd + d] =
              (__bf16)val;
        } else if (mode == 2) {
          const int bb = m / Tt, t = m - bb * Tt;
          const int hh = n / Dd, d = n - hh * Dd;
          ((__bf16*)outp)[(((size_t)bb * Hh + hh) * (size_t)Dd + d) * Tt + t] =
              (__bf16)val;
        } else {
          ((float*)outp)[(size_t)m * N + n] = val;
        }
      }
    }
  }
}

// ---------------------------------------------------------------------------
// Causal flash attention: 1 block = 4 waves = 64 queries of one (b,h).
// Q/K: bf16 [B, H, T, 64]; V: bf16 [B, H, 64, T]; out: bf16 [B, T, H, 64].
// K/V tiles (32 keys) are double-buffered in LDS via async copies.
__global__ __launch_bounds__(128)
void flash_attn_wmma(const __bf16* __restrict__ Qb,
                     const __bf16* __restrict__ Kb,
                     const __bf16* __restrict__ Vt,
                     __bf16* __restrict__ Ob,
                     int Hx, int Tx) {
  constexpr int D = 64;
  __shared__ __bf16 Ktile[2][32 * 64];  // [key][d]
  __shared__ __bf16 Vtile[2][64 * 32];  // [d][key]
  __shared__ __bf16 ptile[4][16 * 32];  // per-wave P round-trip

  const int tid = threadIdx.x;
  const int lane = tid & 31;
  const int wid = tid >> 5;
  const int col = lane & 15, half = lane >> 4;
  const int q0blk = blockIdx.x * 64;
  const int q0 = q0blk + wid * 16;  // this wave's query tile
  const int bh = (int)blockIdx.y;
  const int b = bh / Hx, h = bh - b * Hx;
  const size_t base = (size_t)bh * Tx * D;  // footprint of one (b,h) plane

  // Stage K tile [32 x 64] and V tile [64 x 32] for key block j0 into buf.
  auto stage = [&](int j0, int buf) {
    const __bf16* Kg = Kb + base + (size_t)j0 * D;
    const __bf16* Vg = Vt + base + j0;
#pragma unroll
    for (int p = 0; p < 2; ++p) {
      const int ke = p * 1024 + tid * 8;  // 2048 elements, contiguous
      async_b128(&Ktile[buf][ke], ke * 2, Kg);
      const int row = p * 32 + (tid >> 2);  // 64 rows x 32 keys
      const int cc = (tid & 3) * 8;
      async_b128(&Vtile[buf][row * 32 + cc], (row * Tx + cc) * 2, Vg);
    }
  };

  // Preload Q as two A fragments (HD=64 -> K chunks of 32)
  v16bf qa[2];
#pragma unroll
  for (int c = 0; c < 2; ++c)
    qa[c] = load_frag_a(Qb + base + (size_t)q0 * D + c * 32, D, lane);

  v8f o[4];
#pragma unroll
  for (int s = 0; s < 4; ++s) o[s] = zero8();
  float mrow[8], lrow[8];
#pragma unroll
  for (int v = 0; v < 8; ++v) { mrow[v] = -1e30f; lrow[v] = 0.f; }

  const float scale = 0.125f;      // 1/sqrt(64)
  const int njt = q0blk / 32 + 2;  // causal: key tiles covering <= q0blk+63
  const int qmax = q0 + 15;

  stage(0, 0);
  wait_async0();
  __syncthreads();

  for (int jt = 0; jt < njt; ++jt) {
    const int j0 = jt * 32;
    const int cur = jt & 1;

    if (jt + 1 < njt) stage((jt + 1) * 32, cur ^ 1);  // prefetch next tile

    if (j0 <= qmax) {  // this wave has unmasked keys in the tile
      // S = Q x K^T -> two 16x16 f32 tiles (B fragments from LDS)
      v8f s0 = zero8(), s1 = zero8();
#pragma unroll
      for (int c = 0; c < 2; ++c) {
        const v16bf kf0 = load_frag_b(&Ktile[cur][c * 32], D, lane);
        s0 = wmma_bf16(qa[c], kf0, s0);
        const v16bf kf1 = load_frag_b(&Ktile[cur][16 * 64 + c * 32], D, lane);
        s1 = wmma_bf16(qa[c], kf1, s1);
      }

      // Scale + causal mask + online softmax (16-lane shuffle reductions)
      float alpha[8];
#pragma unroll
      for (int v = 0; v < 8; ++v) {
        const int m = q0 + v + 8 * half;  // global query row
        const float a0 = (j0 + col <= m) ? s0[v] * scale : -1e30f;
        const float a1 = (j0 + 16 + col <= m) ? s1[v] * scale : -1e30f;
        float mx = fmaxf(a0, a1);
#pragma unroll
        for (int off = 1; off < 16; off <<= 1)
          mx = fmaxf(mx, __shfl_xor(mx, off, 32));
        const float mn = fmaxf(mrow[v], mx);
        alpha[v] = __expf(mrow[v] - mn);
        mrow[v] = mn;
        const float p0 = __expf(a0 - mn);
        const float p1 = __expf(a1 - mn);
        float rs = p0 + p1;
#pragma unroll
        for (int off = 1; off < 16; off <<= 1) rs += __shfl_xor(rs, off, 32);
        lrow[v] = lrow[v] * alpha[v] + rs;
        const int lm = v + 8 * half;  // local row
        ptile[wid][lm * 32 + col] = (__bf16)p0;
        ptile[wid][lm * 32 + 16 + col] = (__bf16)p1;
      }

#pragma unroll
      for (int s = 0; s < 4; ++s)
#pragma unroll
        for (int v = 0; v < 8; ++v) o[s][v] *= alpha[v];

      // Same-wave LDS store->load ordering (DS ops are in-order per wave);
      // the asm only pins the compiler's memory ordering.
      asm volatile("" ::: "memory");
      v16bf pa;
      {
        const int row = lane & 15;
        const __bf16* r = &ptile[wid][row * 32];
        const v8bf lo = *(const v8bf*)(r + 8 * half);
        const v8bf hi = *(const v8bf*)(r + 16 + 8 * half);
        pa = __builtin_shufflevector(lo, hi, 0, 1, 2, 3, 4, 5, 6, 7, 8, 9, 10,
                                     11, 12, 13, 14, 15);
      }

      // O += P x V : B(k=key, n=d) = Vtile[d][key]
#pragma unroll
      for (int s = 0; s < 4; ++s) {
        const v16bf vf = load_frag_b(&Vtile[cur][(16 * s) * 32], 32, lane);
        o[s] = wmma_bf16(pa, vf, o[s]);
      }
    }

    // ALL waves: drain this block's async copies, then block barrier.
    wait_async0();
    __syncthreads();
  }

  // Normalize and store to [B, T, H, D] (== row-major [B*T, C])
#pragma unroll
  for (int s = 0; s < 4; ++s) {
    const int d = 16 * s + col;
#pragma unroll
    for (int v = 0; v < 8; ++v) {
      const int t = q0 + v + 8 * half;
      const float val = o[s][v] / lrow[v];
      Ob[(((size_t)b * Tx + t) * Hx + h) * D + d] = (__bf16)val;
    }
  }
}

// ---------------------------------------------------------------------------
extern "C" void kernel_launch(void* const* d_in, const int* in_sizes, int n_in,
                              void* d_out, int out_size, void* d_ws,
                              size_t ws_size, hipStream_t stream) {
  const float* x  = (const float*)d_in[0];
  const float* Wq = (const float*)d_in[1];
  const float* bq = (const float*)d_in[2];
  const float* Wk = (const float*)d_in[3];
  const float* bk = (const float*)d_in[4];
  const float* Wv = (const float*)d_in[5];
  const float* bv = (const float*)d_in[6];
  const float* Wo = (const float*)d_in[7];
  const float* bo = (const float*)d_in[8];

  constexpr int Bn = 2, T = 2048, C = 1024, H = 16, D = 64;
  constexpr int M = Bn * T;  // 4096

  // Workspace carve-out (bf16): x, 4 transposed weights, Q, K, V(t), attn-out
  __bf16* ws = (__bf16*)d_ws;
  size_t off = 0;
  __bf16* xb  = ws + off; off += (size_t)M * C;
  __bf16* Wqt = ws + off; off += (size_t)C * C;
  __bf16* Wkt = ws + off; off += (size_t)C * C;
  __bf16* Wvt = ws + off; off += (size_t)C * C;
  __bf16* Wot = ws + off; off += (size_t)C * C;
  __bf16* Qb  = ws + off; off += (size_t)M * C;
  __bf16* Kb  = ws + off; off += (size_t)M * C;
  __bf16* Vtb = ws + off; off += (size_t)M * C;
  __bf16* Ob  = ws + off; off += (size_t)M * C;

  // 1) precision conversion (+ weight transposition)
  f32_to_bf16_kernel<<<(M * C + 255) / 256, 256, 0, stream>>>(x, xb, M * C);
  transpose_to_bf16_kernel<<<(C * C + 255) / 256, 256, 0, stream>>>(Wq, Wqt, C, C);
  transpose_to_bf16_kernel<<<(C * C + 255) / 256, 256, 0, stream>>>(Wk, Wkt, C, C);
  transpose_to_bf16_kernel<<<(C * C + 255) / 256, 256, 0, stream>>>(Wv, Wvt, C, C);
  transpose_to_bf16_kernel<<<(C * C + 255) / 256, 256, 0, stream>>>(Wo, Wot, C, C);

  // 2) QKV projections: Q,K head-major; V head-transposed
  dim3 gg(M / 32, C / 64);
  wmma_gemm_bf16<<<gg, 32, 0, stream>>>(xb, C, Wqt, C, bq, (void*)Qb, 0,
                                        M, C, C, T, H, D);
  wmma_gemm_bf16<<<gg, 32, 0, stream>>>(xb, C, Wkt, C, bk, (void*)Kb, 0,
                                        M, C, C, T, H, D);
  wmma_gemm_bf16<<<gg, 32, 0, stream>>>(xb, C, Wvt, C, bv, (void*)Vtb, 2,
                                        M, C, C, T, H, D);

  // 3) causal flash attention (async LDS staging) -> bf16 [B, T, H, D]
  dim3 ga(T / 64, Bn * H);
  flash_attn_wmma<<<ga, 128, 0, stream>>>(Qb, Kb, Vtb, Ob, H, T);

  // 4) output projection -> f32 d_out [B*T, C]
  wmma_gemm_bf16<<<gg, 32, 0, stream>>>(Ob, C, Wot, C, bo, d_out, 1,
                                        M, C, C, T, H, D);
}
